// SSMBlock_71416716198225
// MI455X (gfx1250) — matrix operand order repaired and checked by reading, not compiled, not run on hardware
//
#include <hip/hip_runtime.h>
#include <hip/hip_bf16.h>
#include <math.h>

typedef __attribute__((ext_vector_type(16))) __bf16 v16bf;
typedef __attribute__((ext_vector_type(8)))  float  v8f;

struct __align__(4) bf2 { __bf16 x, y; };

#define BATCH   2
#define SEQ     2048
#define D_MODEL 1024
#define D_INNER 2048
#define D_STATE 16
#define D_CONV  4
#define DT_RANK 64
#define ROWS    (BATCH*SEQ)            /* 4096 */
#define DTBC_N  (DT_RANK + 2*D_STATE)  /* 96 */

// ---------------------------------------------------------------------------
// Tiled WMMA GEMM: C[M,N] = act(A[M,K] @ B[K,N] (+ bias))
// 256 threads = 8 wave32s; block tile 64(M)x128(N); wave tile 16x64 (4 accs).
// LDS tiles kept in per-lane FRAGMENT order: each lane's 16 bf16 operand
// values are 32 contiguous aligned bytes -> 2x ds_load_b128 per fragment.
// Staging fully unrolled with hoisted pointers; all global loads issued
// unconditionally (clause-batched, one wait per slab).
// BCHK=0: N divisible by 128, no bounds checks at all.
// BCHK=1: clamp B column index for the load, select 0 at convert; guard store.
// Requires: M % 64 == 0, K % 32 == 0 (true for all call sites).
// MODE 0: plain store.  MODE 1: softplus(acc + bias[col]).
// ---------------------------------------------------------------------------
template<int MODE, int BCHK>
__global__ __launch_bounds__(256) void wmma_gemm(
    float* __restrict__ C, const float* __restrict__ A, const float* __restrict__ B,
    const float* __restrict__ bias, int M, int N, int K, int lda, int ldb, int ldc)
{
  // [tile][lane][j] ; lane = hi*16 + l
  __shared__ __align__(32) __bf16 sAp[4][32][16];   // 4 M-tiles of 16
  __shared__ __align__(32) __bf16 sBp[8][32][16];   // 8 N-tiles of 16

  const int tid   = threadIdx.x;
  const int wave  = tid >> 5;
  const int lane  = tid & 31;
  const int l     = lane & 15;
  const int hi    = lane >> 4;
  const int waveM = wave & 3;      // 4 M-tiles of 16 -> 64
  const int waveN = wave >> 2;     // 2 N-groups of 64 -> 128
  const int blockM = blockIdx.y * 64;
  const int blockN = blockIdx.x * 128;

  // ---- hoisted staging coordinates (k-invariant) -------------------------
  // A: 64x32 slab = 1024 bf16-pairs, 4 per thread (b64 global reads)
  const float* aSrc[4];  __bf16* aDst[4];
  #pragma unroll
  for (int it = 0; it < 4; ++it) {
    const int e  = tid + it * 256;
    const int i  = e >> 4;            // row in tile 0..63
    const int kk = (e & 15) * 2;      // even K 0..30
    aSrc[it] = A + (long)(blockM + i) * lda + kk;
    aDst[it] = &sAp[i >> 4][((kk >> 3) & 1) * 16 + (i & 15)]
                   [(kk & 7) | ((kk & 16) >> 1)];
  }
  // B: 32x128 slab = 2048 bf16-pairs, 8 per thread (2 b32 reads each)
  const float* bSrc[8];  __bf16* bDst[8];  float bMask[8];
  #pragma unroll
  for (int it = 0; it < 8; ++it) {
    const int e  = tid + it * 256;
    const int jc = e & 127;           // column in tile
    const int kk = (e >> 7) * 2;      // even K 0..30
    int gc = blockN + jc;
    if (BCHK) {                       // clamp so load stays unconditional
      bMask[it] = (gc < N) ? 1.f : 0.f;
      gc = (gc < N) ? gc : (N - 1);
    }
    bSrc[it] = B + (long)kk * ldb + gc;
    bDst[it] = &sBp[jc >> 4][((kk >> 3) & 1) * 16 + (jc & 15)][(kk & 7) | ((kk & 16) >> 1)];
  }

  v8f acc0 = {}, acc1 = {}, acc2 = {}, acc3 = {};

  for (int k0 = 0; k0 < K; k0 += 32) {
    // prefetch next A k-slab (global_prefetch_b8)
    if (k0 + 32 < K)
      __builtin_prefetch(aSrc[0] + k0 + 32, 0, 3);

    // ---- issue ALL staging global loads first (clause-batched) -----------
    float2 av[4];
    #pragma unroll
    for (int it = 0; it < 4; ++it)
      av[it] = *(const float2*)(aSrc[it] + k0);
    const long koff = (long)k0 * ldb;
    float bv0[8], bv1[8];
    #pragma unroll
    for (int it = 0; it < 8; ++it) {
      bv0[it] = bSrc[it][koff];
      bv1[it] = bSrc[it][koff + ldb];
    }
    // ---- convert + LDS stores (ds_store_b32 each) ------------------------
    #pragma unroll
    for (int it = 0; it < 4; ++it) {
      bf2 t; t.x = (__bf16)av[it].x; t.y = (__bf16)av[it].y;
      *(bf2*)aDst[it] = t;
    }
    #pragma unroll
    for (int it = 0; it < 8; ++it) {
      float v0 = bv0[it], v1 = bv1[it];
      if (BCHK) { v0 *= bMask[it]; v1 *= bMask[it]; }
      bf2 t; t.x = (__bf16)v0; t.y = (__bf16)v1;
      *(bf2*)bDst[it] = t;
    }
    __syncthreads();

    // ---- fragments: 32 contiguous bytes/lane (2x ds_load_b128 each) ------
    const v16bf af = *(const v16bf*)sAp[waveM][lane];
    const v16bf b0 = *(const v16bf*)sBp[waveN * 4 + 0][lane];
    const v16bf b1 = *(const v16bf*)sBp[waveN * 4 + 1][lane];
    const v16bf b2 = *(const v16bf*)sBp[waveN * 4 + 2][lane];
    const v16bf b3 = *(const v16bf*)sBp[waveN * 4 + 3][lane];

    acc0 = __builtin_amdgcn_wmma_f32_16x16x32_bf16(false, af, false, b0,
                                                   (short)0, acc0, false, false);
    acc1 = __builtin_amdgcn_wmma_f32_16x16x32_bf16(false, af, false, b1,
                                                   (short)0, acc1, false, false);
    acc2 = __builtin_amdgcn_wmma_f32_16x16x32_bf16(false, af, false, b2,
                                                   (short)0, acc2, false, false);
    acc3 = __builtin_amdgcn_wmma_f32_16x16x32_bf16(false, af, false, b3,
                                                   (short)0, acc3, false, false);
    __syncthreads();
  }

  // C/D layout: VGPR i -> row i (lanes 0-15) / row i+8 (lanes 16-31), col = l
  const int rowBase = blockM + waveM * 16 + hi * 8;
  #pragma unroll
  for (int sub = 0; sub < 4; ++sub) {
    const v8f acc = (sub == 0) ? acc0 : (sub == 1) ? acc1 : (sub == 2) ? acc2 : acc3;
    const int gcol = blockN + waveN * 64 + sub * 16 + l;
    if (BCHK && gcol >= N) continue;
    const float bv = (MODE == 1) ? bias[gcol] : 0.f;
    #pragma unroll
    for (int i = 0; i < 8; ++i) {
      float v = acc[i];
      if (MODE == 1) {                    // softplus(acc + bias)
        v += bv;
        v = (v > 20.f) ? v : log1pf(__expf(v));
      }
      C[(long)(rowBase + i) * ldc + gcol] = v;
    }
  }
}

// ---------------------------------------------------------------------------
// Depthwise causal conv (k=4) + bias + silu, elementwise over [ROWS, D_INNER]
// ---------------------------------------------------------------------------
__global__ __launch_bounds__(256) void conv_silu_kernel(
    float* __restrict__ u, const float* __restrict__ u_pre,
    const float* __restrict__ w, const float* __restrict__ b)
{
  long idx = (long)blockIdx.x * 256 + threadIdx.x;
  if (idx >= (long)ROWS * D_INNER) return;
  int  d = (int)(idx % D_INNER);
  long r = idx / D_INNER;
  int  t = (int)(r % SEQ);
  float acc = b[d];
  #pragma unroll
  for (int k = 0; k < D_CONV; ++k) {
    int tt = t - (D_CONV - 1) + k;
    if (tt >= 0)
      acc += w[d * D_CONV + k] * u_pre[(r - (D_CONV - 1) + k) * D_INNER + d];
  }
  u[idx] = acc / (1.f + __expf(-acc));   // silu
}

// ---------------------------------------------------------------------------
// Selective scan. One thread per (batch, channel); 16-state in registers;
// per-timestep B_t/C_t (32 floats) broadcast through LDS.
// g buffer holds z on entry, y*silu(z) on exit (in-place per element).
// ---------------------------------------------------------------------------
__global__ __launch_bounds__(256) void scan_kernel(
    float* __restrict__ g,
    const float* __restrict__ u, const float* __restrict__ dt,
    const float* __restrict__ dtbc, const float* __restrict__ A_log,
    const float* __restrict__ Dp)
{
  __shared__ float sBC[2 * D_STATE];
  const int b = blockIdx.x / (D_INNER / 256);
  const int d = (blockIdx.x % (D_INNER / 256)) * 256 + threadIdx.x;

  float h[D_STATE], Av[D_STATE];
  #pragma unroll
  for (int s = 0; s < D_STATE; ++s) {
    h[s]  = 0.f;
    Av[s] = -__expf(A_log[d * D_STATE + s]);
  }
  const float Dv = Dp[d];

  for (int t = 0; t < SEQ; ++t) {
    const long r = (long)b * SEQ + t;
    __syncthreads();
    if (threadIdx.x < 2 * D_STATE)
      sBC[threadIdx.x] = dtbc[r * DTBC_N + DT_RANK + threadIdx.x];
    __syncthreads();
    const float ut  = u [r * D_INNER + d];
    const float dtt = dt[r * D_INNER + d];
    const float xdt = ut * dtt;
    float y = 0.f;
    #pragma unroll
    for (int s = 0; s < D_STATE; ++s) {
      h[s] = h[s] * __expf(dtt * Av[s]) + xdt * sBC[s];
      y   += h[s] * sBC[D_STATE + s];
    }
    y += ut * Dv;
    const float zv = g[r * D_INNER + d];
    g[r * D_INNER + d] = y * (zv / (1.f + __expf(-zv)));
  }
}

// ---------------------------------------------------------------------------
extern "C" void kernel_launch(void* const* d_in, const int* in_sizes, int n_in,
                              void* d_out, int out_size, void* d_ws, size_t ws_size,
                              hipStream_t stream) {
  const float* x      = (const float*)d_in[0];
  const float* W_in   = (const float*)d_in[1];
  const float* conv_w = (const float*)d_in[2];
  const float* conv_b = (const float*)d_in[3];
  const float* W_x    = (const float*)d_in[4];
  const float* W_dt   = (const float*)d_in[5];
  const float* b_dt   = (const float*)d_in[6];
  const float* A_log  = (const float*)d_in[7];
  const float* D_par  = (const float*)d_in[8];
  const float* W_out  = (const float*)d_in[9];
  float* out = (float*)d_out;

  float* ws = (float*)d_ws;
  const long RD = (long)ROWS * D_INNER;        // 8M floats
  float* u_pre = ws;                            // [ROWS, D_INNER]  (later: dt)
  float* zbuf  = ws + RD;                       // [ROWS, D_INNER]  z -> g
  float* ubuf  = ws + 2 * RD;                   // [ROWS, D_INNER]  conv+silu(u)
  float* dtbc  = ws + 3 * RD;                   // [ROWS, 96]
  float* dtbuf = u_pre;                         // reuse after conv consumed u_pre

  dim3 blk(256);

  // xz = x @ W_in  -> u_pre (cols 0..2047), zbuf (cols 2048..4095)
  {
    dim3 grid(D_INNER / 128, ROWS / 64);
    wmma_gemm<0,0><<<grid, blk, 0, stream>>>(u_pre, x, W_in, nullptr,
        ROWS, D_INNER, D_MODEL, D_MODEL, 2 * D_INNER, D_INNER);
    wmma_gemm<0,0><<<grid, blk, 0, stream>>>(zbuf, x, W_in + D_INNER, nullptr,
        ROWS, D_INNER, D_MODEL, D_MODEL, 2 * D_INNER, D_INNER);
  }

  // u = silu(causal_conv(u_pre) + conv_b)
  conv_silu_kernel<<<dim3((unsigned)((RD + 255) / 256)), blk, 0, stream>>>(
      ubuf, u_pre, conv_w, conv_b);

  // dtbc = u @ W_x   (N = 96 -> bounds-checked variant)
  {
    dim3 grid((DTBC_N + 127) / 128, ROWS / 64);
    wmma_gemm<0,1><<<grid, blk, 0, stream>>>(dtbc, ubuf, W_x, nullptr,
        ROWS, DTBC_N, D_INNER, D_INNER, DTBC_N, DTBC_N);
  }

  // dt = softplus(dtbc[:, :64] @ W_dt + b_dt)
  {
    dim3 grid(D_INNER / 128, ROWS / 64);
    wmma_gemm<1,0><<<grid, blk, 0, stream>>>(dtbuf, dtbc, W_dt, b_dt,
        ROWS, D_INNER, DT_RANK, DTBC_N, D_INNER, D_INNER);
  }

  // selective scan + fused y*silu(z): zbuf becomes g
  scan_kernel<<<dim3(BATCH * (D_INNER / 256)), blk, 0, stream>>>(
      zbuf, ubuf, dtbuf, dtbc, A_log, D_par);

  // out = g @ W_out
  {
    dim3 grid(D_MODEL / 128, ROWS / 64);
    wmma_gemm<0,0><<<grid, blk, 0, stream>>>(out, zbuf, W_out, nullptr,
        ROWS, D_MODEL, D_INNER, D_INNER, D_MODEL, D_MODEL);
  }
}
